// Drug_6365141532844
// MI455X (gfx1250) — matrix operand order repaired and checked by reading, not compiled, not run on hardware
//
#include <hip/hip_runtime.h>
#include <hip/hip_bf16.h>
#include <math.h>

#ifndef USE_ASYNC_LDS
#define USE_ASYNC_LDS 1   // global_load_async_to_lds_b128 + s_wait_asynccnt
#endif

typedef __attribute__((ext_vector_type(16))) _Float16 v16h;
typedef __attribute__((ext_vector_type(8)))  _Float16 v8h;
typedef __attribute__((ext_vector_type(4)))  _Float16 v4h;
typedef __attribute__((ext_vector_type(2)))  _Float16 v2h;
typedef __attribute__((ext_vector_type(8)))  float    v8f;

#define NN 8000
#define EN 64000
#define FN 64
#define HN 10
#define ET (EN + NN)
#define NHF ((size_t)NN * HN * FN)
#define INV_SQRT_F 0.125f

// ---------------------------------------------------------------------------
// WMMA fragment helpers (v_wmma_f32_16x16x32_f16), layouts per CDNA5 ISA 7.12.2
// A 16x32 f16: lane(m=l&15, kb=(l>>4)*8) holds K=[k0+kb,+8) and [k0+kb+16,+8)
//              -> two contiguous 16B chunks of a row-major row.
// B 32x16 f16: lane(n=l&15, hi=l>>4) holds K=k0+hi*8+t (t<8), k0+16+hi*8+t-8 (t>=8) -> we pre-
//              swizzle W so these 16 halves are contiguous (32B) per lane.
// C/D 16x16 f32: lane: N = l&15, VGPR j -> M = (l>>4)*8 + j.
// ---------------------------------------------------------------------------
static __device__ __forceinline__ v16h cat8(v8h lo, v8h hi) {
  return __builtin_shufflevector(lo, hi, 0,1,2,3,4,5,6,7,8,9,10,11,12,13,14,15);
}

// A fragment from a row-major [16 x 64] tile (global or LDS pointer).
static __device__ __forceinline__ v16h lda_rm(const _Float16* tile, int lane, int k0) {
  int m  = lane & 15;
  int kb = (lane >> 4) << 3;
  const _Float16* p = tile + m * 64 + k0 + kb;
  v8h lo = *(const v8h*)(p);
  v8h hi = *(const v8h*)(p + 16);
  return cat8(lo, hi);
}

// B fragment from fragment-major swizzled weights: frag = nb*2+ks.
static __device__ __forceinline__ v16h ldb_swz(const _Float16* wswz, int lane, int frag) {
  const _Float16* p = wswz + (((frag << 5) + lane) << 4);
  v8h lo = *(const v8h*)(p);
  v8h hi = *(const v8h*)(p + 8);
  return cat8(lo, hi);
}

static __device__ __forceinline__ v8f wmma32(v16h a, v16h b, v8f c) {
  return __builtin_amdgcn_wmma_f32_16x16x32_f16(false, a, false, b, (short)0, c, false, false);
}

// ---------------------------------------------------------------------------
// Elementwise / setup kernels
// ---------------------------------------------------------------------------
__global__ void k_fill(float* p, float v, int n) {
  int i = blockIdx.x * blockDim.x + threadIdx.x;
  if (i < n) p[i] = v;
}

// 4-wide f32 -> f16 convert (n multiple of 4)
__global__ void k_cvt4(const float* __restrict__ s, _Float16* __restrict__ d, int n4) {
  int i = blockIdx.x * blockDim.x + threadIdx.x;
  if (i >= n4) return;
  float4 f = *(const float4*)(s + (size_t)i * 4);
  v4h h; h[0] = (_Float16)f.x; h[1] = (_Float16)f.y; h[2] = (_Float16)f.z; h[3] = (_Float16)f.w;
  *(v4h*)(d + (size_t)i * 4) = h;
}

// Convert + swizzle a [64 x ncols] f32 weight into fragment-major f16 blocks:
// dst[cg*4096 + ((nb*2+ks)*32 + lane)*16 + t] = W[(ks*32 + (lane>>4)*8 + (t&7) + (t>>3)*16)*ncols + cg*64 + (nb*16 + (lane&15))]
__global__ void k_swz(const float* __restrict__ W, _Float16* __restrict__ dst,
                      int ncols, int total) {
  int i = blockIdx.x * blockDim.x + threadIdx.x;
  if (i >= total) return;
  int cg   = i >> 12;
  int r    = i & 4095;
  int frag = r >> 9;
  int lane = (r >> 4) & 31;
  int t    = r & 15;
  int nb = frag >> 1, ks = frag & 1;
  int k = ks * 32 + ((lane >> 4) << 3) + (t & 7) + ((t >> 3) << 4);
  int n = cg * 64 + (nb << 4) + (lane & 15);
  dst[i] = (_Float16)W[(size_t)k * ncols + n];
}

// deg[dst]+=1, cnt[src]+=1, loop_sum[dst,f]+=edge_attr[e,f]
__global__ void k_degsum(const float* __restrict__ eattr, const int* __restrict__ src,
                         const int* __restrict__ dst, float* deg, float* cnt, float* lsum) {
  int t = blockIdx.x * blockDim.x + threadIdx.x;
  if (t >= EN * FN) return;
  int e = t >> 6, f = t & 63;
  int d = dst[e];
  atomicAdd(&lsum[(size_t)d * FN + f], eattr[t]);
  if (f == 0) { atomicAdd(&deg[d], 1.0f); atomicAdd(&cnt[src[e]], 1.0f); }
}

// loop_attr -> f16 rows E..E+N-1 of ea16
__global__ void k_loopattr(const float* __restrict__ lsum, const float* __restrict__ deg,
                           _Float16* __restrict__ ea16) {
  int t = blockIdx.x * blockDim.x + threadIdx.x;
  if (t >= NN * FN) return;
  int n = t >> 6;
  float la = lsum[t] / fmaxf(deg[n], 1.0f);
  ea16[(size_t)(EN + n) * FN + (t & 63)] = (_Float16)la;
}

// ---------------------------------------------------------------------------
// WMMA GEMM, LDS-free: C[M,Ncols] = A16[M,64] @ Wswz[64,Ncols]
// 128 threads = 4 waves; one 16-row tile per wave; blockIdx.y = 64-col group.
// MODE 0: f32 out  1: f32+f16 out  2: f16 out  3: atomic segment-sum by source
// ---------------------------------------------------------------------------
template <int MODE>
__global__ __launch_bounds__(128) void k_gemm(
    const _Float16* __restrict__ A, const _Float16* __restrict__ Wswz,
    float* __restrict__ Cf, _Float16* __restrict__ Ch,
    float* __restrict__ Seg, const int* __restrict__ srcIdx, int Ncols) {
  int wid = threadIdx.x >> 5, lane = threadIdx.x & 31;
  int cg = blockIdx.y;
  int row0 = (blockIdx.x * 4 + wid) * 16;
  const _Float16* Arow = A + (size_t)row0 * 64;
  const _Float16* Wf = Wswz + (size_t)cg * 4096;

  v16h a0 = lda_rm(Arow, lane, 0);
  v16h a1 = lda_rm(Arow, lane, 32);
  v8f acc[4];
#pragma unroll
  for (int nb = 0; nb < 4; ++nb) {
    v8f c = {};
    c = wmma32(a0, ldb_swz(Wf, lane, nb * 2 + 0), c);
    c = wmma32(a1, ldb_swz(Wf, lane, nb * 2 + 1), c);
    acc[nb] = c;
  }

  int nn = lane & 15, mhi = (lane >> 4) << 3;
#pragma unroll
  for (int nb = 0; nb < 4; ++nb) {
#pragma unroll
    for (int j = 0; j < 8; ++j) {
      int row = row0 + mhi + j;
      int col = cg * 64 + nb * 16 + nn;
      float v = acc[nb][j];
      if (MODE == 0) {
        Cf[(size_t)row * Ncols + col] = v;
      } else if (MODE == 1) {
        Cf[(size_t)row * Ncols + col] = v;
        Ch[(size_t)row * Ncols + col] = (_Float16)v;
      } else if (MODE == 2) {
        Ch[(size_t)row * Ncols + col] = (_Float16)v;
      } else {
        int sr = (row < EN) ? srcIdx[row] : (row - EN);
        atomicAdd(&Seg[(size_t)sr * 64 + col], v);
      }
    }
  }
}

// ---------------------------------------------------------------------------
// Attention: one wave per node.
// ---------------------------------------------------------------------------
__global__ __launch_bounds__(256) void k_attn(
    const float* __restrict__ XS, const float* __restrict__ XQ,
    const float* __restrict__ XK, const float* __restrict__ XV,
    const float* __restrict__ SEK, const float* __restrict__ SEV,
    const float* __restrict__ SEQ, const float* __restrict__ cnt,
    const float* __restrict__ bias, float* __restrict__ node_out,
    _Float16* __restrict__ NU16) {
  int wid = threadIdx.x >> 5, lane = threadIdx.x & 31;
  int n = blockIdx.x * 8 + wid;
  if (n >= NN) return;
  int c0 = lane * 2;
  size_t nb64 = (size_t)n * 64 + c0;
  float2 sek = *(const float2*)(SEK + nb64);
  float2 sev = *(const float2*)(SEV + nb64);
  float2 seq = *(const float2*)(SEQ + nb64);
  float cn = cnt[n];

  float qk[HN], vs0[HN], vs1[HN];
#pragma unroll
  for (int h = 0; h < HN; ++h) {
    size_t base = ((size_t)n * HN + h) * 64 + c0;
    float2 kv = *(const float2*)(XK + base);
    float2 vv = *(const float2*)(XV + base);
    float2 qv = *(const float2*)(XQ + base);
    float ks0 = cn * kv.x + sek.x, ks1 = cn * kv.y + sek.y;
    vs0[h] = cn * vv.x + sev.x;
    vs1[h] = cn * vv.y + sev.y;
    float p = (qv.x + seq.x) * ks0 + (qv.y + seq.y) * ks1;
#pragma unroll
    for (int m = 16; m >= 1; m >>= 1) p += __shfl_xor(p, m, 32);
    qk[h] = p * INV_SQRT_F;
  }
  float mx = qk[0];
#pragma unroll
  for (int h = 1; h < HN; ++h) mx = fmaxf(mx, qk[h]);
  float al[HN], sum = 0.f;
#pragma unroll
  for (int h = 0; h < HN; ++h) { al[h] = __expf(qk[h] - mx); sum += al[h]; }
  float inv = 1.0f / sum;
#pragma unroll
  for (int h = 0; h < HN; ++h) {
    float a = al[h] * inv;
    size_t base = ((size_t)n * HN + h) * 64 + c0;
    float2 xs = *(const float2*)(XS + base);
    float2 bi = *(const float2*)(bias + h * 64 + c0);
    float nu0 = xs.x * a * vs0[h];
    float nu1 = xs.y * a * vs1[h];
    float2 no; no.x = nu0 + bi.x; no.y = nu1 + bi.y;
    *(float2*)(node_out + base) = no;
    v2h nh; nh[0] = (_Float16)nu0; nh[1] = (_Float16)nu1;
    *(v2h*)(NU16 + base) = nh;
  }
}

// ---------------------------------------------------------------------------
// LayerNorm directly on WMMA C fragments; row sums via 16-lane xor butterflies.
// ---------------------------------------------------------------------------
static __device__ __forceinline__ void ln_frag(
    const v8f acc[4], const float* ea, const float* g, const float* bv,
    int lane, float out[4][8]) {
  int nn = lane & 15;
  int mhi = (lane >> 4) << 3;
#pragma unroll
  for (int j = 0; j < 8; ++j) {
    float v[4], s = 0.f, s2 = 0.f;
#pragma unroll
    for (int nb = 0; nb < 4; ++nb) {
      float t = acc[nb][j] + ea[(mhi + j) * 64 + nb * 16 + nn];
      v[nb] = t; s += t; s2 += t * t;
    }
#pragma unroll
    for (int msk = 8; msk >= 1; msk >>= 1) {
      s  += __shfl_xor(s,  msk, 32);
      s2 += __shfl_xor(s2, msk, 32);
    }
    float mu  = s * 0.015625f;
    float var = s2 * 0.015625f - mu * mu;
    float rs  = rsqrtf(var);
#pragma unroll
    for (int nb = 0; nb < 4; ++nb) {
      int col = nb * 16 + nn;
      out[nb][j] = (v[nb] - mu) * rs * g[col] + bv[col];
    }
  }
}

// ---------------------------------------------------------------------------
// Fused edge branch: block = 5 waves = 8 edges x 10 heads = five 16x64 tiles.
// W5 (swizzled) + per-wave edge_attr tiles staged via async global->LDS copies.
// ---------------------------------------------------------------------------
__global__ __launch_bounds__(160) void k_edge(
    const _Float16* __restrict__ A16, const _Float16* __restrict__ B16,
    const float* __restrict__ eattr, const _Float16* __restrict__ W5swz,
    const int* __restrict__ srcIdx,
    const float* __restrict__ g1v, const float* __restrict__ b1v,
    const float* __restrict__ g2v, const float* __restrict__ b2v,
    float* __restrict__ edge_out) {
  __shared__ _Float16 W5s[64 * 64];      // swizzled fragment-major
  __shared__ _Float16 Msh[5][16 * 64];   // row-major m / u tiles
  __shared__ float EAsh[5][16 * 64];     // row-major edge_attr tiles
  __shared__ float Acc[8 * 64];
  __shared__ float Gs[4 * 64];
  int tid = threadIdx.x, wid = tid >> 5, lane = tid & 31;
  int eBase = blockIdx.x * 8;
  int r0 = eBase * HN + wid * 16;

#if USE_ASYNC_LDS
  // stage swizzled W5 (512 x 16B chunks, block-wide)
  for (int ch = tid; ch < 512; ch += 160) {
    unsigned lds = (unsigned)(size_t)&W5s[ch * 8];
    unsigned long long g = (unsigned long long)(size_t)(W5swz + ch * 8);
    asm volatile("global_load_async_to_lds_b128 %0, %1, off" :: "v"(lds), "v"(g) : "memory");
  }
  // stage this wave's 16x64 f32 edge_attr tile (256 x 16B chunks, 8 per lane)
#pragma unroll
  for (int it = 0; it < 8; ++it) {
    int ch = lane + it * 32;            // 0..255
    int m = ch >> 4, q = ch & 15;       // row, 16B quarter
    int e = (r0 + m) / HN;
    unsigned lds = (unsigned)(size_t)&EAsh[wid][m * 64 + q * 4];
    unsigned long long g = (unsigned long long)(size_t)(eattr + (size_t)e * 64 + q * 4);
    asm volatile("global_load_async_to_lds_b128 %0, %1, off" :: "v"(lds), "v"(g) : "memory");
  }
#else
  for (int i = tid; i < 4096; i += 160) W5s[i] = W5swz[i];
#endif

  if (tid < 64) { Gs[tid] = g1v[tid]; Gs[64+tid] = b1v[tid]; Gs[128+tid] = g2v[tid]; Gs[192+tid] = b2v[tid]; }
  for (int i = tid; i < 512; i += 160) Acc[i] = 0.f;

  // build m = relu(A[s_e,h] + B[e]) tile (row-major f16)
  int c = lane * 2;
#pragma unroll
  for (int m = 0; m < 16; ++m) {
    int r = r0 + m;
    int e = r / HN;
    int h = r - e * HN;
    int se = srcIdx[e];
    v2h av = *(const v2h*)(A16 + ((size_t)se * HN + h) * 64 + c);
    v2h bv = *(const v2h*)(B16 + (size_t)e * 64 + c);
    float m0 = fmaxf((float)av[0] + (float)bv[0], 0.f);
    float m1 = fmaxf((float)av[1] + (float)bv[1], 0.f);
    v2h mv; mv[0] = (_Float16)m0; mv[1] = (_Float16)m1;
    *(v2h*)(&Msh[wid][m * 64 + c]) = mv;
#if !USE_ASYNC_LDS
    float2 ev = *(const float2*)(eattr + (size_t)e * 64 + c);
    *(float2*)(&EAsh[wid][m * 64 + c]) = ev;
#endif
  }
#if USE_ASYNC_LDS
  asm volatile("s_wait_asynccnt 0x0" ::: "memory");
#endif
  __syncthreads();

  int nn = lane & 15, mhi = (lane >> 4) << 3;
  float u[4][8];
  {  // pass 1: m @ W5 -> LN(g1,b1)
    v16h a0 = lda_rm(Msh[wid], lane, 0);
    v16h a1 = lda_rm(Msh[wid], lane, 32);
    v8f acc[4];
#pragma unroll
    for (int nb = 0; nb < 4; ++nb) {
      v8f cc = {};
      cc = wmma32(a0, ldb_swz(W5s, lane, nb * 2 + 0), cc);
      cc = wmma32(a1, ldb_swz(W5s, lane, nb * 2 + 1), cc);
      acc[nb] = cc;
    }
    ln_frag(acc, EAsh[wid], &Gs[0], &Gs[64], lane, u);
  }
  // write u back as f16 (same-wave DS ordering)
#pragma unroll
  for (int nb = 0; nb < 4; ++nb)
#pragma unroll
    for (int j = 0; j < 8; ++j)
      Msh[wid][(mhi + j) * 64 + nb * 16 + nn] = (_Float16)u[nb][j];

  float eu[4][8];
  {  // pass 2: u @ W5 -> LN(g2,b2)
    v16h a0 = lda_rm(Msh[wid], lane, 0);
    v16h a1 = lda_rm(Msh[wid], lane, 32);
    v8f acc[4];
#pragma unroll
    for (int nb = 0; nb < 4; ++nb) {
      v8f cc = {};
      cc = wmma32(a0, ldb_swz(W5s, lane, nb * 2 + 0), cc);
      cc = wmma32(a1, ldb_swz(W5s, lane, nb * 2 + 1), cc);
      acc[nb] = cc;
    }
    ln_frag(acc, EAsh[wid], &Gs[128], &Gs[192], lane, eu);
  }
  // mean over heads via LDS accumulation
#pragma unroll
  for (int nb = 0; nb < 4; ++nb)
#pragma unroll
    for (int j = 0; j < 8; ++j) {
      int el = (r0 + mhi + j) / HN - eBase;
      atomicAdd(&Acc[el * 64 + nb * 16 + nn], eu[nb][j]);
    }
  __syncthreads();
  for (int i = tid; i < 128; i += 160) {
    float4 a4 = *(const float4*)(Acc + i * 4);
    float4 o;
    o.x = a4.x * (1.0f/HN); o.y = a4.y * (1.0f/HN);
    o.z = a4.z * (1.0f/HN); o.w = a4.w * (1.0f/HN);
    *(float4*)(edge_out + (size_t)eBase * 64 + i * 4) = o;
  }
}

// ---------------------------------------------------------------------------
extern "C" void kernel_launch(void* const* d_in, const int* in_sizes, int n_in,
                              void* d_out, int out_size, void* d_ws, size_t ws_size,
                              hipStream_t stream) {
  (void)in_sizes; (void)n_in; (void)out_size; (void)ws_size;
  const float* x      = (const float*)d_in[0];
  const int*   ei     = (const int*)d_in[1];
  const int*   src    = ei;
  const int*   dst    = ei + EN;
  const float* eattr  = (const float*)d_in[2];
  const float* W_src  = (const float*)d_in[3];
  const float* Wnq    = (const float*)d_in[4];
  const float* Wnk    = (const float*)d_in[5];
  const float* Wnv    = (const float*)d_in[6];
  const float* Weq    = (const float*)d_in[7];
  const float* Wev    = (const float*)d_in[8];
  const float* Wek    = (const float*)d_in[9];
  const float* W4     = (const float*)d_in[10];
  const float* W5     = (const float*)d_in[11];
  const float* g1v    = (const float*)d_in[12];
  const float* b1v    = (const float*)d_in[13];
  const float* g2v    = (const float*)d_in[14];
  const float* b2v    = (const float*)d_in[15];
  const float* bias   = (const float*)d_in[16];

  float* node_out = (float*)d_out;
  float* edge_out = node_out + NHF;

  char* p = (char*)d_ws;
  auto carve = [&](size_t bytes) { char* r = p; p += (bytes + 255) & ~(size_t)255; return r; };
  float* deg  = (float*)carve((size_t)NN * 4);
  float* cnt  = (float*)carve((size_t)NN * 4);
  float* lsum = (float*)carve((size_t)NN * FN * 4);
  float* XS   = (float*)carve(NHF * 4);
  float* XQ   = (float*)carve(NHF * 4);
  float* XK   = (float*)carve(NHF * 4);
  float* XV   = (float*)carve(NHF * 4);
  float* SEK  = (float*)carve((size_t)NN * FN * 4);
  float* SEV  = (float*)carve((size_t)NN * FN * 4);
  float* SEQ  = (float*)carve((size_t)NN * FN * 4);
  _Float16* x16    = (_Float16*)carve((size_t)NN * FN * 2);
  _Float16* ea16   = (_Float16*)carve((size_t)ET * FN * 2);
  _Float16* XS16   = (_Float16*)carve(NHF * 2);
  _Float16* NU16   = (_Float16*)carve(NHF * 2);
  _Float16* A16    = (_Float16*)carve(NHF * 2);
  _Float16* B16    = (_Float16*)carve((size_t)EN * FN * 2);
  _Float16* Wsrc16 = (_Float16*)carve((size_t)64 * 640 * 2);  // swizzled
  _Float16* Wnq16  = (_Float16*)carve(4096 * 2);
  _Float16* Wnk16  = (_Float16*)carve(4096 * 2);
  _Float16* Wnv16  = (_Float16*)carve(4096 * 2);
  _Float16* Weq16  = (_Float16*)carve(4096 * 2);
  _Float16* Wev16  = (_Float16*)carve(4096 * 2);
  _Float16* Wek16  = (_Float16*)carve(4096 * 2);
  _Float16* W4t16  = (_Float16*)carve(4096 * 2);
  _Float16* W4b16  = (_Float16*)carve(4096 * 2);
  _Float16* W5_16  = (_Float16*)carve(4096 * 2);

  const int T = 256;
  k_fill<<<(NN + T - 1) / T, T, 0, stream>>>(deg, 0.f, NN);
  k_fill<<<(NN + T - 1) / T, T, 0, stream>>>(cnt, 1.f, NN);  // +1 self loop
  k_fill<<<(NN * FN + T - 1) / T, T, 0, stream>>>(lsum, 0.f, NN * FN);
  k_fill<<<(NN * FN + T - 1) / T, T, 0, stream>>>(SEK, 0.f, NN * FN);
  k_fill<<<(NN * FN + T - 1) / T, T, 0, stream>>>(SEV, 0.f, NN * FN);
  k_fill<<<(NN * FN + T - 1) / T, T, 0, stream>>>(SEQ, 0.f, NN * FN);

  k_cvt4<<<(NN * FN / 4 + T - 1) / T, T, 0, stream>>>(x, x16, NN * FN / 4);
  k_cvt4<<<(EN * FN / 4 + T - 1) / T, T, 0, stream>>>(eattr, ea16, EN * FN / 4);

  // weights: convert + fragment-major swizzle
  k_swz<<<(40960 + T - 1) / T, T, 0, stream>>>(W_src, Wsrc16, 640, 40960);
  k_swz<<<(4096 + T - 1) / T, T, 0, stream>>>(Wnq, Wnq16, 64, 4096);
  k_swz<<<(4096 + T - 1) / T, T, 0, stream>>>(Wnk, Wnk16, 64, 4096);
  k_swz<<<(4096 + T - 1) / T, T, 0, stream>>>(Wnv, Wnv16, 64, 4096);
  k_swz<<<(4096 + T - 1) / T, T, 0, stream>>>(Weq, Weq16, 64, 4096);
  k_swz<<<(4096 + T - 1) / T, T, 0, stream>>>(Wev, Wev16, 64, 4096);
  k_swz<<<(4096 + T - 1) / T, T, 0, stream>>>(Wek, Wek16, 64, 4096);
  k_swz<<<(4096 + T - 1) / T, T, 0, stream>>>(W4, W4t16, 64, 4096);            // rows 0..63
  k_swz<<<(4096 + T - 1) / T, T, 0, stream>>>(W4 + 64 * 64, W4b16, 64, 4096);  // rows 64..127
  k_swz<<<(4096 + T - 1) / T, T, 0, stream>>>(W5, W5_16, 64, 4096);

  k_degsum<<<(EN * FN + T - 1) / T, T, 0, stream>>>(eattr, src, dst, deg, cnt, lsum);
  k_loopattr<<<(NN * FN + T - 1) / T, T, 0, stream>>>(lsum, deg, ea16);

  // x_src = x @ W_src (f32 + f16 copy)
  k_gemm<1><<<dim3(125, 10), 128, 0, stream>>>(x16, Wsrc16, XS, XS16, nullptr, nullptr, 640);
  // per-node-head projections [80000,64] @ [64,64]
  k_gemm<0><<<dim3(1250, 1), 128, 0, stream>>>(XS16, Wnq16, XQ, nullptr, nullptr, nullptr, 64);
  k_gemm<0><<<dim3(1250, 1), 128, 0, stream>>>(XS16, Wnk16, XK, nullptr, nullptr, nullptr, 64);
  k_gemm<0><<<dim3(1250, 1), 128, 0, stream>>>(XS16, Wnv16, XV, nullptr, nullptr, nullptr, 64);
  // edge projections with fused segment-sum over source index [72000,64]
  k_gemm<3><<<dim3(1125, 1), 128, 0, stream>>>(ea16, Weq16, nullptr, nullptr, SEK, src, 64);
  k_gemm<3><<<dim3(1125, 1), 128, 0, stream>>>(ea16, Wev16, nullptr, nullptr, SEV, src, 64);
  k_gemm<3><<<dim3(1125, 1), 128, 0, stream>>>(ea16, Wek16, nullptr, nullptr, SEQ, src, 64);

  k_attn<<<dim3(1000, 1), 256, 0, stream>>>(XS, XQ, XK, XV, SEK, SEV, SEQ, cnt, bias,
                                            node_out, NU16);

  // A = node_update @ W4[64:], B = ea @ W4[:64]
  k_gemm<2><<<dim3(1250, 1), 128, 0, stream>>>(NU16, W4b16, nullptr, A16, nullptr, nullptr, 64);
  k_gemm<2><<<dim3(1000, 1), 128, 0, stream>>>(ea16, W4t16, nullptr, B16, nullptr, nullptr, 64);

  k_edge<<<dim3(8000, 1), 160, 0, stream>>>(A16, B16, eattr, W5_16, src,
                                            g1v, b1v, g2v, b2v, edge_out);
}
